// LSTMDecoder_83004537963092
// MI455X (gfx1250) — matrix operand order repaired and checked by reading, not compiled, not run on hardware
//
#include <hip/hip_runtime.h>
#include <hip/hip_bf16.h>

// ---------------------------------------------------------------------------
// LSTM decoder (teacher forcing) for MI455X / gfx1250, wave32 + WMMA.
//
// Factorization: xproj[b,t] = xpre[b] + embW[prev_tok[b,t]] + b_ih + b_hh
//   xpre = x @ W_ih[:, :512]^T            (WMMA GEMM, 256x512x4096)
//   embW = emb @ W_ih[:, 512:768]^T       (WMMA GEMM, 512x256x4096)
// Recurrence (dominant, 64 x 2.15 GFLOP): gacc_t = h_t @ W_hh^T (WMMA GEMM),
//   then elementwise LSTM cell. W_hh f16 copy (8MB) stays L2-resident.
// Output: logits = hs @ W_out^T (WMMA GEMM 16384x1024x512), then per-row
//   log-softmax NLL + deterministic fixed-order tree reduction.
//
// GEMM: 128x128 block tile, 8 waves as 4(M)x2(N), each wave 32x64 (8 WMMA
// accum tiles). A/B K-slices (128x32 f16) staged to LDS with
// GLOBAL_LOAD_ASYNC_TO_LDS_B128 (ASYNCcnt), double-buffered.
// ---------------------------------------------------------------------------

#define B_   256
#define T_   64
#define XD_  512
#define ED_  256
#define DIN_ 768
#define H_   1024
#define G4_  4096   // 4*H
#define NT_  512    // N_TAG

#define LDS_STRIDE 40   // halves per staged row (80B: 16B-aligned, 20-bank stride)

typedef _Float16 v8h  __attribute__((ext_vector_type(8)));
typedef _Float16 v16h __attribute__((ext_vector_type(16)));
typedef float    v8f  __attribute__((ext_vector_type(8)));

// Async global->LDS 16B copy (per-lane). The low 32 bits of a generic pointer
// to __shared__ are the LDS byte offset.
__device__ __forceinline__ void async_ld_b128(const _Float16* gptr, void* lptr) {
  unsigned ldsa = (unsigned)(unsigned long long)lptr;
  unsigned long long ga = (unsigned long long)gptr;
  asm volatile("global_load_async_to_lds_b128 %0, %1, off"
               :: "v"(ldsa), "v"(ga) : "memory");
}
__device__ __forceinline__ void wait_async0() {
  asm volatile("s_wait_asynccnt 0x0" ::: "memory");
}

// ---------------- f32 -> f16 (optionally strided rows) ----------------------
__global__ __launch_bounds__(256)
void cvt_f16_kernel(const float* __restrict__ src, _Float16* __restrict__ dst,
                    int rows, int cols, int src_stride, int src_off) {
  int i = blockIdx.x * 256 + threadIdx.x;
  int n = rows * cols;
  if (i >= n) return;
  int r = i / cols, c = i - r * cols;
  dst[i] = (_Float16)src[(size_t)r * src_stride + src_off + c];
}

// ---------------- state init: c = c0, h16 = f16(h0) -------------------------
__global__ __launch_bounds__(256)
void init_state_kernel(const float* __restrict__ h0, const float* __restrict__ c0,
                       float* __restrict__ c_state, _Float16* __restrict__ h16) {
  int i = blockIdx.x * 256 + threadIdx.x;
  if (i >= B_ * H_) return;
  c_state[i] = c0[i];
  h16[i] = (_Float16)h0[i];
}

// ---------------- WMMA GEMM: C[M,N] = A[M,K] * B[N,K]^T ---------------------
// A, B f16 row-major (K contiguous). Block tile 128x128; M,N multiples of 128,
// K multiple of 32. Grid = (M/128)*(N/128) blocks of 256 threads.
__global__ __launch_bounds__(256)
void wmma_gemm_nt_kernel(const _Float16* __restrict__ A,
                         const _Float16* __restrict__ B,
                         float* __restrict__ C, int M, int N, int K) {
  __shared__ _Float16 sA[2][128 * LDS_STRIDE];
  __shared__ _Float16 sB[2][128 * LDS_STRIDE];

  const int tid  = threadIdx.x;
  const int lane = tid & 31;
  const int wave = tid >> 5;
  const int tiles_n = N >> 7;
  const int bm = blockIdx.x / tiles_n;
  const int bn = blockIdx.x - bm * tiles_n;
  const int mbase = bm << 7;
  const int nbase = bn << 7;

  // Staging: 128 rows x 32 halves (64B) per operand per K-step = 512 x 16B
  // chunks; each thread moves 2 A chunks + 2 B chunks.
  const int ar0 = tid >> 2, as0 = (tid & 3) << 3;   // row, half-offset
  const int ar1 = ar0 + 64;
  const _Float16* gA0 = A + (size_t)(mbase + ar0) * K + as0;
  const _Float16* gA1 = A + (size_t)(mbase + ar1) * K + as0;
  const _Float16* gB0 = B + (size_t)(nbase + ar0) * K + as0;
  const _Float16* gB1 = B + (size_t)(nbase + ar1) * K + as0;
  const int lofs0 = ar0 * LDS_STRIDE + as0;
  const int lofs1 = ar1 * LDS_STRIDE + as0;

  auto issue = [&](int buf, int kb) {
    async_ld_b128(gA0 + kb, &sA[buf][lofs0]);
    async_ld_b128(gA1 + kb, &sA[buf][lofs1]);
    async_ld_b128(gB0 + kb, &sB[buf][lofs0]);
    async_ld_b128(gB1 + kb, &sB[buf][lofs1]);
  };

  // Wave tile: 32(M) x 64(N) at (wm*32, wn*64) within the block tile.
  const int wm = wave >> 1;
  const int wn = wave & 1;
  const int m0 = wm << 5;
  const int n0 = wn << 6;
  const int l15 = lane & 15;
  const int hi  = lane >> 4;          // 0 or 1
  const int kgA = hi << 3;            // A frag: lanes 16-31 hold K+8.. / K+24..
  const int kgB = hi << 3;            // B frag: lanes 16-31 hold K+8.. / K+24..

  v8f acc[2][4] = {};
  issue(0, 0);

  for (int kb = 0; kb < K; kb += 32) {
    const int cur = (kb >> 5) & 1;
    wait_async0();
    __syncthreads();
    if (kb + 32 < K) issue(cur ^ 1, kb + 32);

    const _Float16* tA = &sA[cur][0];
    const _Float16* tB = &sB[cur][0];
    v16h af[2], bf[4];
#pragma unroll
    for (int i = 0; i < 2; ++i) {
      const _Float16* p = tA + (m0 + (i << 4) + l15) * LDS_STRIDE;
      v8h x0 = *(const v8h*)(p + kgA);
      v8h x1 = *(const v8h*)(p + kgA + 16);
#pragma unroll
      for (int q = 0; q < 8; ++q) { af[i][q] = x0[q]; af[i][q + 8] = x1[q]; }
    }
#pragma unroll
    for (int j = 0; j < 4; ++j) {
      const _Float16* p = tB + (n0 + (j << 4) + l15) * LDS_STRIDE + kgB;
      v8h x0 = *(const v8h*)(p);
      v8h x1 = *(const v8h*)(p + 16);
#pragma unroll
      for (int q = 0; q < 8; ++q) { bf[j][q] = x0[q]; bf[j][q + 8] = x1[q]; }
    }
#pragma unroll
    for (int i = 0; i < 2; ++i)
#pragma unroll
      for (int j = 0; j < 4; ++j)
        acc[i][j] = __builtin_amdgcn_wmma_f32_16x16x32_f16(
            false, af[i], false, bf[j], (short)0, acc[i][j], false, false);
  }

  // C/D layout: VGPR v -> row M = v + 8*hi, low lane nibble -> column.
  const int r_off = hi << 3;
#pragma unroll
  for (int i = 0; i < 2; ++i)
#pragma unroll
    for (int j = 0; j < 4; ++j) {
      float* Cb = C + (size_t)(mbase + m0 + (i << 4)) * N + (nbase + n0 + (j << 4));
#pragma unroll
      for (int v = 0; v < 8; ++v)
        Cb[(size_t)(v + r_off) * N + l15] = acc[i][j][v];
    }
}

// ---------------- LSTM cell (elementwise, one step) -------------------------
__device__ __forceinline__ float sigm(float x) { return 1.0f / (1.0f + expf(-x)); }

__global__ __launch_bounds__(256)
void lstm_cell_kernel(const float* __restrict__ gacc,   // [B,4H] = h@W_hh^T
                      const float* __restrict__ xpre,   // [B,4H]
                      const float* __restrict__ embW,   // [N_TAG,4H]
                      const int*   __restrict__ atom_truth, // [B,T]
                      const float* __restrict__ b_ih,
                      const float* __restrict__ b_hh,
                      float* __restrict__ c_state,      // [B,H]
                      _Float16* __restrict__ h16,       // [B,H]
                      _Float16* __restrict__ hs16,      // [T,B,H]
                      int t) {
  int i = blockIdx.x * 256 + threadIdx.x;
  if (i >= B_ * H_) return;
  int b = i >> 10;            // H_ = 1024
  int j = i & (H_ - 1);
  int tok = (t == 0) ? 0 : atom_truth[b * T_ + t - 1];
  size_t gb = (size_t)b * G4_;
  size_t eb = (size_t)tok * G4_;

  float gi = gacc[gb + j]          + xpre[gb + j]          + embW[eb + j]          + b_ih[j]          + b_hh[j];
  float gf = gacc[gb + H_ + j]     + xpre[gb + H_ + j]     + embW[eb + H_ + j]     + b_ih[H_ + j]     + b_hh[H_ + j];
  float gg = gacc[gb + 2 * H_ + j] + xpre[gb + 2 * H_ + j] + embW[eb + 2 * H_ + j] + b_ih[2 * H_ + j] + b_hh[2 * H_ + j];
  float go = gacc[gb + 3 * H_ + j] + xpre[gb + 3 * H_ + j] + embW[eb + 3 * H_ + j] + b_ih[3 * H_ + j] + b_hh[3 * H_ + j];

  float ig = sigm(gi), fg = sigm(gf), gg2 = tanhf(gg), og = sigm(go);
  float c = fg * c_state[i] + ig * gg2;
  c_state[i] = c;
  float h = og * tanhf(c);
  h16[i] = (_Float16)h;
  hs16[(size_t)t * (B_ * H_) + i] = (_Float16)h;
}

// ---------------- per-row log-softmax NLL -----------------------------------
__global__ __launch_bounds__(256)
void nll_row_kernel(const float* __restrict__ logits,    // [T*B, NT] (no bias)
                    const float* __restrict__ b_out,     // [NT]
                    const int*   __restrict__ atom_truth,// [B,T]
                    const int*   __restrict__ atom_mask, // [B,T] (nonzero=true)
                    float* __restrict__ rownll,          // [T*B]
                    float* __restrict__ rowmask) {       // [T*B]
  __shared__ float red[256];
  int r = blockIdx.x;                  // row index in [0, T*B)
  int t = r >> 8;                      // B_ = 256
  int b = r & (B_ - 1);
  int tid = threadIdx.x;
  const float* lrow = logits + (size_t)r * NT_;

  float mx = -3.4e38f;
#pragma unroll
  for (int n = tid; n < NT_; n += 256) {
    float v = lrow[n] + b_out[n];
    mx = fmaxf(mx, v);
  }
  red[tid] = mx; __syncthreads();
  for (int s = 128; s > 0; s >>= 1) {
    if (tid < s) red[tid] = fmaxf(red[tid], red[tid + s]);
    __syncthreads();
  }
  mx = red[0]; __syncthreads();

  float se = 0.0f;
#pragma unroll
  for (int n = tid; n < NT_; n += 256) {
    float v = lrow[n] + b_out[n];
    se += expf(v - mx);
  }
  red[tid] = se; __syncthreads();
  for (int s = 128; s > 0; s >>= 1) {
    if (tid < s) red[tid] += red[tid + s];
    __syncthreads();
  }

  if (tid == 0) {
    float lse = mx + logf(red[0]);
    int tgt = atom_truth[b * T_ + t];
    float tv = lrow[tgt] + b_out[tgt];
    float m = (atom_mask[b * T_ + t] != 0) ? 1.0f : 0.0f;
    rownll[r] = (lse - tv) * m;
    rowmask[r] = m;
  }
}

// ---------------- deterministic final reduction -----------------------------
__global__ __launch_bounds__(256)
void final_reduce_kernel(const float* __restrict__ rownll,
                         const float* __restrict__ rowmask,
                         float* __restrict__ out) {
  __shared__ float sn[256];
  __shared__ float sm[256];
  int tid = threadIdx.x;
  float an = 0.0f, am = 0.0f;
  for (int r = tid; r < T_ * B_; r += 256) { an += rownll[r]; am += rowmask[r]; }
  sn[tid] = an; sm[tid] = am; __syncthreads();
  for (int s = 128; s > 0; s >>= 1) {
    if (tid < s) { sn[tid] += sn[tid + s]; sm[tid] += sm[tid + s]; }
    __syncthreads();
  }
  if (tid == 0) out[0] = sn[0] / fmaxf(sm[0], 1.0f);
}

// ---------------------------------------------------------------------------
extern "C" void kernel_launch(void* const* d_in, const int* in_sizes, int n_in,
                              void* d_out, int out_size, void* d_ws, size_t ws_size,
                              hipStream_t stream) {
  // setup_inputs() order:
  const float* x          = (const float*)d_in[0];   // [B, 512]
  const int*   atom_truth = (const int*)  d_in[1];   // [B, T]
  const int*   atom_mask  = (const int*)  d_in[2];   // [B, T] bool->int
  const float* emb        = (const float*)d_in[3];   // [512, 256]
  const float* W_ih       = (const float*)d_in[4];   // [4096, 768]
  const float* W_hh       = (const float*)d_in[5];   // [4096, 1024]
  const float* b_ih       = (const float*)d_in[6];   // [4096]
  const float* b_hh       = (const float*)d_in[7];   // [4096]
  const float* W_out      = (const float*)d_in[8];   // [512, 1024]
  const float* b_out      = (const float*)d_in[9];   // [512]
  const float* h0         = (const float*)d_in[10];  // [B, H]
  const float* c0         = (const float*)d_in[11];  // [B, H]
  float* out = (float*)d_out;

  // ---- workspace carve-out (256B aligned) ----
  char* w = (char*)d_ws;
  size_t off = 0;
  auto alloc = [&](size_t bytes) -> char* {
    off = (off + 255) & ~(size_t)255;
    char* p = w + off;
    off += bytes;
    return p;
  };
  _Float16* Whh16  = (_Float16*)alloc((size_t)G4_ * H_ * 2);   // 8 MiB
  _Float16* WihX16 = (_Float16*)alloc((size_t)G4_ * XD_ * 2);  // 4 MiB
  _Float16* WihE16 = (_Float16*)alloc((size_t)G4_ * ED_ * 2);  // 2 MiB
  _Float16* Wout16 = (_Float16*)alloc((size_t)NT_ * H_ * 2);   // 1 MiB
  _Float16* x16    = (_Float16*)alloc((size_t)B_ * XD_ * 2);
  _Float16* emb16  = (_Float16*)alloc((size_t)NT_ * ED_ * 2);
  float*    xpre   = (float*)   alloc((size_t)B_ * G4_ * 4);   // 4 MiB
  float*    embW   = (float*)   alloc((size_t)NT_ * G4_ * 4);  // 8 MiB
  float*    gacc   = (float*)   alloc((size_t)B_ * G4_ * 4);   // 4 MiB
  float*    c_st   = (float*)   alloc((size_t)B_ * H_ * 4);
  _Float16* h16    = (_Float16*)alloc((size_t)B_ * H_ * 2);
  _Float16* hs16   = (_Float16*)alloc((size_t)T_ * B_ * H_ * 2);   // 32 MiB
  float*    logits = (float*)   alloc((size_t)T_ * B_ * NT_ * 4);  // 32 MiB
  float*    rownll = (float*)   alloc((size_t)T_ * B_ * 4);
  float*    rowmsk = (float*)   alloc((size_t)T_ * B_ * 4);
  (void)ws_size; (void)n_in; (void)in_sizes; (void)out_size;

  auto blks = [](int n) { return (n + 255) / 256; };

  // ---- f16 conversions ----
  cvt_f16_kernel<<<blks(G4_ * H_), 256, 0, stream>>>(W_hh, Whh16, G4_, H_, H_, 0);
  cvt_f16_kernel<<<blks(G4_ * XD_), 256, 0, stream>>>(W_ih, WihX16, G4_, XD_, DIN_, 0);
  cvt_f16_kernel<<<blks(G4_ * ED_), 256, 0, stream>>>(W_ih, WihE16, G4_, ED_, DIN_, XD_);
  cvt_f16_kernel<<<blks(NT_ * H_), 256, 0, stream>>>(W_out, Wout16, NT_, H_, H_, 0);
  cvt_f16_kernel<<<blks(B_ * XD_), 256, 0, stream>>>(x, x16, B_, XD_, XD_, 0);
  cvt_f16_kernel<<<blks(NT_ * ED_), 256, 0, stream>>>(emb, emb16, NT_, ED_, ED_, 0);
  init_state_kernel<<<blks(B_ * H_), 256, 0, stream>>>(h0, c0, c_st, h16);

  // ---- precompute xpre = x16 @ WihX16^T, embW = emb16 @ WihE16^T ----
  wmma_gemm_nt_kernel<<<(B_ / 128) * (G4_ / 128), 256, 0, stream>>>(
      x16, WihX16, xpre, B_, G4_, XD_);
  wmma_gemm_nt_kernel<<<(NT_ / 128) * (G4_ / 128), 256, 0, stream>>>(
      emb16, WihE16, embW, NT_, G4_, ED_);

  // ---- recurrence: 64 x (WMMA GEMM + cell) ----
  for (int t = 0; t < T_; ++t) {
    wmma_gemm_nt_kernel<<<(B_ / 128) * (G4_ / 128), 256, 0, stream>>>(
        h16, Whh16, gacc, B_, G4_, H_);
    lstm_cell_kernel<<<blks(B_ * H_), 256, 0, stream>>>(
        gacc, xpre, embW, atom_truth, b_ih, b_hh, c_st, h16, hs16, t);
  }

  // ---- output projection: logits = hs16 @ Wout16^T ----
  wmma_gemm_nt_kernel<<<(T_ * B_ / 128) * (NT_ / 128), 256, 0, stream>>>(
      hs16, Wout16, logits, T_ * B_, NT_, H_);

  // ---- masked NLL ----
  nll_row_kernel<<<T_ * B_, 256, 0, stream>>>(logits, b_out, atom_truth,
                                              atom_mask, rownll, rowmsk);
  final_reduce_kernel<<<1, 256, 0, stream>>>(rownll, rowmsk, out);
}
